// BondMessagePassing_36309653521074
// MI455X (gfx1250) — matrix operand order repaired and checked by reading, not compiled, not run on hardware
//
#include <hip/hip_runtime.h>
#include <hip/hip_bf16.h>
#include <math.h>

typedef __attribute__((ext_vector_type(16))) __bf16 v16bf;
typedef __attribute__((ext_vector_type(8)))  __bf16 v8bf;
typedef __attribute__((ext_vector_type(8)))  float   v8f;

#define NUM_NODES 1024
#define E_EDGES   4096
#define HID       256
#define BOND      64
#define HEADS     8
#define HD        32

__device__ __forceinline__ float gelu_exact(float x) {
    return 0.5f * x * (1.0f + erff(x * 0.70710678118654752f));
}

__device__ __forceinline__ v8f wmma_bf16(v16bf a, v16bf b, v8f c) {
    return __builtin_amdgcn_wmma_f32_16x16x32_bf16(
        false, a, false, b, (short)0, c, false, false);
}

// ---- async global -> LDS copy (gfx1250, ASYNCcnt-tracked) -----------------
__device__ __forceinline__ void async_load_b128(void* lds_dst, const void* gsrc) {
    unsigned ldsa = (unsigned)(uintptr_t)lds_dst;  // low 32 bits = LDS byte addr
    asm volatile("global_load_async_to_lds_b128 %0, %1, off"
                 :: "v"(ldsa), "v"(gsrc) : "memory");
}
// wait until at most N async ops remain outstanding (in-order completion)
__device__ __forceinline__ void wait_async_le2() {
    asm volatile("s_wait_asynccnt 0x2" ::: "memory");
}
__device__ __forceinline__ void wait_async0() {
    asm volatile("s_wait_asynccnt 0x0" ::: "memory");
}

// ---- 16-lane xor-butterfly reductions (pure VALU on gfx1250) --------------
#if __has_builtin(__builtin_amdgcn_permlane16)
__device__ __forceinline__ float plane16(float v, unsigned lo, unsigned hi) {
    int x = __builtin_amdgcn_permlane16(__float_as_int(v), __float_as_int(v),
                                        lo, hi, false, false);
    return __int_as_float(x);
}
__device__ __forceinline__ float red16_max(float v) {
    v = fmaxf(v, plane16(v, 0x67452301u, 0xEFCDAB89u));  // xor 1
    v = fmaxf(v, plane16(v, 0x54761032u, 0xDCFE98BAu));  // xor 2
    v = fmaxf(v, plane16(v, 0x32107654u, 0xBA98FEDCu));  // xor 4
    v = fmaxf(v, plane16(v, 0xFEDCBA98u, 0x76543210u));  // xor 8
    return v;
}
__device__ __forceinline__ float red16_sum(float v) {
    v += plane16(v, 0x67452301u, 0xEFCDAB89u);
    v += plane16(v, 0x54761032u, 0xDCFE98BAu);
    v += plane16(v, 0x32107654u, 0xBA98FEDCu);
    v += plane16(v, 0xFEDCBA98u, 0x76543210u);
    return v;
}
#else
__device__ __forceinline__ float red16_max(float v) {
    for (int s = 1; s < 16; s <<= 1) v = fmaxf(v, __shfl_xor(v, s, 32));
    return v;
}
__device__ __forceinline__ float red16_sum(float v) {
    for (int s = 1; s < 16; s <<= 1) v += __shfl_xor(v, s, 32);
    return v;
}
#endif

// ---- WMMA fragment loaders (wave32, 16x16x32 bf16) ------------------------
__device__ __forceinline__ v16bf load_frag_a(const __bf16* base, int ld) {
    const int lane = threadIdx.x & 31;
    const int m = lane & 15, g = lane >> 4;
    const __bf16* p = base + m * ld;
    v8bf lo = *(const v8bf*)(p + 8 * g);       // K = 8g .. 8g+7
    v8bf hi = *(const v8bf*)(p + 16 + 8 * g);  // K = 16+8g .. 23+8g
    return __builtin_shufflevector(lo, hi, 0,1,2,3,4,5,6,7,8,9,10,11,12,13,14,15);
}
__device__ __forceinline__ v16bf load_frag_b(const __bf16* base, int ld) {
    const int lane = threadIdx.x & 31;
    const int n = lane & 15, g = lane >> 4;
    const __bf16* p = base + n * ld;
    v8bf lo = *(const v8bf*)(p + 8 * g);
    v8bf hi = *(const v8bf*)(p + 16 + 8 * g);
    return __builtin_shufflevector(lo, hi, 0,1,2,3,4,5,6,7,8,9,10,11,12,13,14,15);
}

// ---- Generic GEMM: out[M,N] = epi(A[M,K] @ W[K,N] + bias) -----------------
// EPI: 0 = bias, 1 = gelu(bias), 2 = bias + 2*R.  OUTB: 1 -> bf16 output.
#define TM 64
#define TN 128
#define TK 32
#define LDA_P 40
#define LDB_P 40

template<int EPI, int OUTB>
__global__ __launch_bounds__(256)
void gemm_kernel(const float* __restrict__ A, const float* __restrict__ W,
                 const float* __restrict__ bias, const float* __restrict__ R,
                 float* __restrict__ out, __bf16* __restrict__ outb,
                 int M, int N, int K)
{
    __shared__ alignas(16) __bf16 sA[TM * LDA_P];
    __shared__ alignas(16) __bf16 sB[TN * LDB_P];
    const int bm = blockIdx.x, bn = blockIdx.y;
    const int tid = threadIdx.x;
    const int wave = tid >> 5;
    const int m_sub = wave & 3;
    const int n_half = wave >> 2;
    v8f acc[4] = {};

    for (int k0 = 0; k0 < K; k0 += TK) {
        for (int idx = tid; idx < TM * TK; idx += 256) {
            int r = idx >> 5, c = idx & 31;
            sA[r * LDA_P + c] = (__bf16)A[(size_t)(bm * TM + r) * K + k0 + c];
        }
        for (int idx = tid; idx < TN * TK; idx += 256) {
            int n = idx & 127, c = idx >> 7;
            sB[n * LDB_P + c] = (__bf16)W[(size_t)(k0 + c) * N + bn * TN + n];
        }
        __syncthreads();
        v16bf a = load_frag_a(sA + m_sub * 16 * LDA_P, LDA_P);
#pragma unroll
        for (int j = 0; j < 4; ++j) {
            v16bf b = load_frag_b(sB + (n_half * 64 + j * 16) * LDB_P, LDB_P);
            acc[j] = wmma_bf16(a, b, acc[j]);
        }
        __syncthreads();
    }
    const int lane = tid & 31;
    const int cn = lane & 15, g = lane >> 4;
#pragma unroll
    for (int j = 0; j < 4; ++j) {
        int col = bn * TN + n_half * 64 + j * 16 + cn;
#pragma unroll
        for (int i = 0; i < 8; ++i) {
            int row = bm * TM + m_sub * 16 + i + 8 * g;
            float v = acc[j][i] + bias[col];
            if (EPI == 1) v = gelu_exact(v);
            if (EPI == 2) v += 2.0f * R[(size_t)row * N + col];
            if (OUTB) outb[(size_t)row * N + col] = (__bf16)v;
            else      out [(size_t)row * N + col] = v;
        }
    }
}

// ---- Flash attention, double-buffered async pipeline ----------------------
// qkv is bf16 [E,768]. K tiles: async global->LDS into sK[2] (next block's
// loads stay in flight behind the current block's WMMAs; s_wait_asynccnt 0x2
// releases only the older pair). V tiles: register-prefetched b128 loads,
// scattered to a transposed LDS tile at the top of each iteration.
__global__ __launch_bounds__(128)
void attn_kernel(const __bf16* __restrict__ qkv, float* __restrict__ o)
{
    __shared__ alignas(16) __bf16 sQ[64 * 40];
    __shared__ alignas(16) __bf16 sK[2][64 * 40];
    __shared__ alignas(16) __bf16 sVt[32 * 72];     // V transposed [dim][key]
    __shared__ alignas(16) __bf16 sP[4 * 16 * 72];  // per-wave P staging
    const int head = blockIdx.y, qb = blockIdx.x;
    const int tid = threadIdx.x, wave = tid >> 5, lane = tid & 31;
    const int cn = lane & 15, g = lane >> 4;
    const float scale = 0.17677669529663687f;  // 1/sqrt(32)

    // this thread's two staging chunks: (row r, 8-elem col chunk c)
    const int r0 = tid >> 2,         c0 = (tid & 3) * 8;
    const int r1 = (tid + 128) >> 2, c1 = ((tid + 128) & 3) * 8;

    // ---- prologue: Q + K(0) async, V(0) into registers ----
    async_load_b128(sQ + r0 * 40 + c0,
                    qkv + (size_t)(qb * 64 + r0) * 768 + head * HD + c0);
    async_load_b128(sQ + r1 * 40 + c1,
                    qkv + (size_t)(qb * 64 + r1) * 768 + head * HD + c1);
    async_load_b128(sK[0] + r0 * 40 + c0,
                    qkv + (size_t)r0 * 768 + 256 + head * HD + c0);
    async_load_b128(sK[0] + r1 * 40 + c1,
                    qkv + (size_t)r1 * 768 + 256 + head * HD + c1);
    v8bf vv0 = *(const v8bf*)(qkv + (size_t)r0 * 768 + 512 + head * HD + c0);
    v8bf vv1 = *(const v8bf*)(qkv + (size_t)r1 * 768 + 512 + head * HD + c1);
    wait_async0();
    __syncthreads();
    v16bf qf = load_frag_a(sQ + wave * 16 * 40, 40);

    float mrow[8], lrow[8];
#pragma unroll
    for (int i = 0; i < 8; ++i) { mrow[i] = -1e30f; lrow[i] = 0.f; }
    v8f oacc[2] = {};

    int p = 0;
    for (int kb = 0; kb < E_EDGES; kb += 64) {
        __syncthreads();  // prior compute done: safe to overwrite sVt / sK[p^1]
        // scatter current V block (in registers) into transposed LDS tile
#pragma unroll
        for (int u = 0; u < 8; ++u) sVt[(c0 + u) * 72 + r0] = vv0[u];
#pragma unroll
        for (int u = 0; u < 8; ++u) sVt[(c1 + u) * 72 + r1] = vv1[u];
        if (kb + 64 < E_EDGES) {
            // issue next block's K async copies + V register prefetch
            async_load_b128(sK[p ^ 1] + r0 * 40 + c0,
                            qkv + (size_t)(kb + 64 + r0) * 768 + 256 + head * HD + c0);
            async_load_b128(sK[p ^ 1] + r1 * 40 + c1,
                            qkv + (size_t)(kb + 64 + r1) * 768 + 256 + head * HD + c1);
            vv0 = *(const v8bf*)(qkv + (size_t)(kb + 64 + r0) * 768 + 512 + head * HD + c0);
            vv1 = *(const v8bf*)(qkv + (size_t)(kb + 64 + r1) * 768 + 512 + head * HD + c1);
            wait_async_le2();  // current block's K landed; next block's in flight
        } else {
            wait_async0();
        }
        __syncthreads();
        const __bf16* kbuf = sK[p];
        // S = Q(16x32) @ K^T -> 16x64
        v8f sc[4] = {};
#pragma unroll
        for (int j = 0; j < 4; ++j) {
            v16bf kf = load_frag_b(kbuf + j * 16 * 40, 40);
            sc[j] = wmma_bf16(qf, kf, sc[j]);
        }
        // online softmax (scale folded into exp arg); row i+8g <-> vgpr slot i
#pragma unroll
        for (int i = 0; i < 8; ++i) {
            float rm = sc[0][i];
#pragma unroll
            for (int j = 1; j < 4; ++j) rm = fmaxf(rm, sc[j][i]);
            rm = red16_max(rm);
            float mnew = fmaxf(mrow[i], rm * scale);
            float a = __expf(mrow[i] - mnew);
            mrow[i] = mnew;
            float rs = 0.f;
#pragma unroll
            for (int j = 0; j < 4; ++j) {
                float pj = __expf(fmaf(sc[j][i], scale, -mnew));
                sc[j][i] = pj; rs += pj;
            }
            rs = red16_sum(rs);
            lrow[i] = lrow[i] * a + rs;
            oacc[0][i] *= a;
            oacc[1][i] *= a;
        }
        // stage P (16x64) per wave for A-fragment reuse
        __bf16* pw = sP + wave * 16 * 72;
#pragma unroll
        for (int j = 0; j < 4; ++j)
#pragma unroll
            for (int i = 0; i < 8; ++i)
                pw[(i + 8 * g) * 72 + j * 16 + cn] = (__bf16)sc[j][i];
        // O += P(16x64) @ V(64x32)
#pragma unroll
        for (int kk = 0; kk < 2; ++kk) {
            v16bf pf = load_frag_a(pw + kk * 32, 72);
#pragma unroll
            for (int j = 0; j < 2; ++j) {
                v16bf vf = load_frag_b(sVt + j * 16 * 72 + kk * 32, 72);
                oacc[j] = wmma_bf16(pf, vf, oacc[j]);
            }
        }
        p ^= 1;
    }
#pragma unroll
    for (int j = 0; j < 2; ++j)
#pragma unroll
        for (int i = 0; i < 8; ++i) {
            int row = qb * 64 + wave * 16 + i + 8 * g;
            int col = head * HD + j * 16 + cn;
            o[(size_t)row * HID + col] = oacc[j][i] / lrow[i];
        }
}

// ---- aggregation / layernorm helpers --------------------------------------
__global__ __launch_bounds__(256)
void zero_kernel(float* __restrict__ p, int n) {
    int i = blockIdx.x * 256 + threadIdx.x;
    if (i < n) p[i] = 0.f;
}

__global__ __launch_bounds__(256)
void scatter_kernel(const int* __restrict__ tgt, const float* __restrict__ h,
                    float* __restrict__ S, float* __restrict__ deg) {
    const int e = blockIdx.x, c = threadIdx.x;
    const int t = tgt[e];
    unsafeAtomicAdd(&S[(size_t)t * HID + c], h[(size_t)e * HID + c]);
    if (c == 0) unsafeAtomicAdd(&deg[t], 1.0f);
}

__global__ __launch_bounds__(256)
void gather_kernel(const int* __restrict__ tgt, const float* __restrict__ h,
                   const float* __restrict__ S, const float* __restrict__ deg,
                   float* __restrict__ r) {
    const int e = blockIdx.x, c = threadIdx.x;
    const int t = tgt[e];
    r[(size_t)e * HID + c] = S[(size_t)t * HID + c] - deg[t] * h[(size_t)e * HID + c];
}

__global__ __launch_bounds__(256)
void ln_kernel(const float* __restrict__ x, const float* __restrict__ gm,
               const float* __restrict__ bt, float* __restrict__ y, int do_gelu) {
    __shared__ float red[8];
    const int row = blockIdx.x, c = threadIdx.x;
    float v = x[(size_t)row * HID + c];
    float s = v;
#pragma unroll
    for (int i = 1; i < 32; i <<= 1) s += __shfl_xor(s, i, 32);
    if ((c & 31) == 0) red[c >> 5] = s;
    __syncthreads();
    float mean = 0.f;
#pragma unroll
    for (int i = 0; i < 8; ++i) mean += red[i];
    mean *= (1.0f / HID);
    __syncthreads();
    float d = v - mean, sq = d * d;
#pragma unroll
    for (int i = 1; i < 32; i <<= 1) sq += __shfl_xor(sq, i, 32);
    if ((c & 31) == 0) red[c >> 5] = sq;
    __syncthreads();
    float var = 0.f;
#pragma unroll
    for (int i = 0; i < 8; ++i) var += red[i];
    var *= (1.0f / HID);
    float o = d * rsqrtf(var + 1e-5f) * gm[c] + bt[c];
    if (do_gelu) o = gelu_exact(o);
    y[(size_t)row * HID + c] = o;
}

// ---- host orchestration ----------------------------------------------------
extern "C" void kernel_launch(void* const* d_in, const int* in_sizes, int n_in,
                              void* d_out, int out_size, void* d_ws, size_t ws_size,
                              hipStream_t stream) {
    const int*   edge_index = (const int*)d_in[0];
    const int*   tgt   = edge_index + E_EDGES;  // row 1 of [2,E]
    const float* bond  = (const float*)d_in[1];
    const float* W_emb = (const float*)d_in[2];
    const float* b_emb = (const float*)d_in[3];
    const float* W_h   = (const float*)d_in[4];
    const float* b_h   = (const float*)d_in[5];
    const float* ln1_g = (const float*)d_in[6];
    const float* ln1_b = (const float*)d_in[7];
    const float* in_w  = (const float*)d_in[8];
    const float* in_b  = (const float*)d_in[9];
    const float* out_w = (const float*)d_in[10];
    const float* out_b = (const float*)d_in[11];
    const float* up_w  = (const float*)d_in[12];
    const float* up_b  = (const float*)d_in[13];
    const float* ln2_g = (const float*)d_in[14];
    const float* ln2_b = (const float*)d_in[15];
    float* out = (float*)d_out;

    float* ws   = (float*)d_ws;
    float* h    = ws;                       // E*H f32
    float* r    = h   + E_EDGES * HID;      // E*H f32 (also embed tmp)
    float* xn   = r   + E_EDGES * HID;      // E*H f32 (also u)
    float* qkvf = xn  + E_EDGES * HID;      // E*3H region, used as bf16
    float* ao   = qkvf + E_EDGES * 3 * HID; // E*H f32
    float* tb   = ao  + E_EDGES * HID;      // E*H f32
    float* S    = tb  + E_EDGES * HID;      // N*H f32
    __bf16* qkvB = (__bf16*)qkvf;           // E*3H bf16 (fits in f32 region)
    float*  deg  = S + NUM_NODES * HID;     // zeroed together with S

    dim3 blk256(256), blk128(128);
    dim3 g256(E_EDGES / TM, HID / TN);
    dim3 g768(E_EDGES / TM, (3 * HID) / TN);

    // h = gelu(bond @ W_emb + b_emb) @ W_h + b_h
    gemm_kernel<1, 0><<<g256, blk256, 0, stream>>>(bond, W_emb, b_emb, nullptr,
                                                   r, nullptr, E_EDGES, HID, BOND);
    gemm_kernel<0, 0><<<g256, blk256, 0, stream>>>(r, W_h, b_h, nullptr,
                                                   h, nullptr, E_EDGES, HID, HID);

    for (int t = 0; t < 3; ++t) {
        int nz = NUM_NODES * HID + NUM_NODES;
        zero_kernel<<<(nz + 255) / 256, blk256, 0, stream>>>(S, nz);
        scatter_kernel<<<E_EDGES, blk256, 0, stream>>>(tgt, h, S, deg);
        gather_kernel<<<E_EDGES, blk256, 0, stream>>>(tgt, h, S, deg, r);
        ln_kernel<<<E_EDGES, blk256, 0, stream>>>(r, ln1_g + t * HID, ln1_b + t * HID, xn, 0);
        // qkv in bf16 so attention can async-copy straight to LDS
        gemm_kernel<0, 1><<<g768, blk256, 0, stream>>>(xn, in_w + (size_t)t * HID * 3 * HID,
                                                       in_b + t * 3 * HID, nullptr,
                                                       nullptr, qkvB, E_EDGES, 3 * HID, HID);
        attn_kernel<<<dim3(E_EDGES / 64, HEADS), blk128, 0, stream>>>(qkvB, ao);
        gemm_kernel<2, 0><<<g256, blk256, 0, stream>>>(ao, out_w + (size_t)t * HID * HID,
                                                       out_b + t * HID, r,
                                                       tb, nullptr, E_EDGES, HID, HID);
        gemm_kernel<0, 0><<<g256, blk256, 0, stream>>>(tb, up_w + (size_t)t * HID * HID,
                                                       up_b + t * HID, nullptr,
                                                       xn, nullptr, E_EDGES, HID, HID);
        float* hout = (t == 2) ? out : h;
        ln_kernel<<<E_EDGES, blk256, 0, stream>>>(xn, ln2_g + t * HID, ln2_b + t * HID, hout, 1);
    }
}
